// PatchSelectNetwork_30623116821346
// MI455X (gfx1250) — matrix operand order repaired and checked by reading, not compiled, not run on hardware
//
#include <hip/hip_runtime.h>
#include <cstdint>

// ---------------------------------------------------------------------------
// PatchSelectNetwork on MI455X (gfx1250, wave32, WMMA bf16 -> f32 accum)
// GEMMs: LDS double-buffered async-copy pipeline + 32x32 register tiles/wave.
// ---------------------------------------------------------------------------

#define HID    768
#define HEADS  12
#define HD     64
#define MLPD   3072
#define TXT    256
#define IMG    1024
#define SEQ    1280
#define NPATCH 16
#define SELN   4
#define NITER  4

typedef __attribute__((ext_vector_type(16))) __bf16 v16bf;
typedef __attribute__((ext_vector_type(8)))  float  v8f;

union FragU { uint4 u[2]; v16bf v; };

__device__ __forceinline__ unsigned short f2bf(float f) {
  unsigned int u = __float_as_uint(f);
  unsigned int r = 0x7FFFu + ((u >> 16) & 1u);   // round-to-nearest-even
  return (unsigned short)((u + r) >> 16);
}

__device__ __forceinline__ float wave_sum(float v) {
  v += __shfl_xor(v, 16, 32);
  v += __shfl_xor(v, 8, 32);
  v += __shfl_xor(v, 4, 32);
  v += __shfl_xor(v, 2, 32);
  v += __shfl_xor(v, 1, 32);
  return v;
}

// LDS byte offset of a __shared__ object: generic (flat) LDS pointers carry the
// wave-relative LDS offset in their low 32 bits (addr[31:0] truncation rule).
__device__ __forceinline__ unsigned lds_off(const void* p) {
  return (unsigned)(uintptr_t)p;
}

// Async copy 16B from global to LDS (ASYNCcnt-tracked; per-lane addresses).
__device__ __forceinline__ void async_b128(unsigned laddr, const void* gaddr) {
  asm volatile("global_load_async_to_lds_b128 %0, %1, off"
               :: "v"(laddr), "v"(gaddr) : "memory");
}
__device__ __forceinline__ void wait_async0() {
  asm volatile("s_wait_asynccnt 0" ::: "memory");
}

// ---------------------------------------------------------------------------
// Tiled GEMM: C[M,N] = A[M,K](bf16) @ B, with B pre-transposed Bt[N,K](bf16).
// Requires M%64==0, N%128==0, K%32==0.
// Block: 256 threads = 8 waves; block tile 64(M) x 128(N).
// Wave grid 2(M) x 4(N); each wave: 2 A-frags x 2 B-frags -> 4 WMMA per k-step.
// A/B k-tiles double-buffered in LDS, filled by async-to-LDS copies.
// ---------------------------------------------------------------------------
__global__ void k_gemm_bf16_lds(const unsigned short* __restrict__ A,
                                const unsigned short* __restrict__ Bt,
                                const float* __restrict__ bias,
                                float* __restrict__ C, int M, int N, int K) {
  __shared__ __align__(16) unsigned short sA[2][64 * 32];    //  8 KB
  __shared__ __align__(16) unsigned short sB[2][128 * 32];   // 16 KB

  const int tid  = threadIdx.x;
  const int lane = tid & 31;
  const int wid  = tid >> 5;
  const int wy   = wid >> 2;           // 0..1  (M)
  const int wx   = wid & 3;            // 0..3  (N)
  const int r    = lane & 15;
  const int hi   = lane >> 4;

  const int tm = blockIdx.y * 64;      // block row base
  const int tn = blockIdx.x * 128;     // block col base

  // staging addresses owned by this thread
  const int arow = tid >> 2, ac4 = (tid & 3) * 8;                 // A: 1 chunk
  const int brow0 = tid >> 2, brow1 = 64 + (tid >> 2);            // B: 2 chunks
  const int bc4 = (tid & 3) * 8;
  const unsigned short* gA = A  + (size_t)(tm + arow) * K + ac4;
  const unsigned short* gB0 = Bt + (size_t)(tn + brow0) * K + bc4;
  const unsigned short* gB1 = Bt + (size_t)(tn + brow1) * K + bc4;

  // prologue: stage k=0 into buffer 0
  async_b128(lds_off(&sA[0][arow * 32 + ac4]),  gA);
  async_b128(lds_off(&sB[0][brow0 * 32 + bc4]), gB0);
  async_b128(lds_off(&sB[0][brow1 * 32 + bc4]), gB1);
  wait_async0();
  __syncthreads();

  v8f acc00 = {}, acc01 = {}, acc10 = {}, acc11 = {};
  const int nk = K >> 5;
  for (int kt = 0; kt < nk; ++kt) {
    const int buf = kt & 1;
    if (kt + 1 < nk) {                 // stage k+1 into the other buffer
      const int nb = buf ^ 1;
      const int k1 = (kt + 1) << 5;
      async_b128(lds_off(&sA[nb][arow * 32 + ac4]),  gA  + k1);
      async_b128(lds_off(&sB[nb][brow0 * 32 + bc4]), gB0 + k1);
      async_b128(lds_off(&sB[nb][brow1 * 32 + bc4]), gB1 + k1);
    }
    // fragments from LDS (ds_load_b128)
    FragU a0, a1, b0, b1;
    {
      const unsigned short* p = &sA[buf][(wy * 32 + r) * 32];
      a0.u[0] = *(const uint4*)(p + hi * 8);
      a0.u[1] = *(const uint4*)(p + 16 + hi * 8);
      p += 16 * 32;
      a1.u[0] = *(const uint4*)(p + hi * 8);
      a1.u[1] = *(const uint4*)(p + 16 + hi * 8);
    }
    {
      const unsigned short* p = &sB[buf][(wx * 32 + r) * 32] + hi * 8;
      b0.u[0] = ((const uint4*)p)[0];
      b0.u[1] = ((const uint4*)p)[2];
      p += 16 * 32;
      b1.u[0] = ((const uint4*)p)[0];
      b1.u[1] = ((const uint4*)p)[2];
    }
    // 4 WMMAs; A fragments reused across the two N fragments
    acc00 = __builtin_amdgcn_wmma_f32_16x16x32_bf16(false, a0.v, false, b0.v,
                                                    (short)0, acc00, false, false);
    acc01 = __builtin_amdgcn_wmma_f32_16x16x32_bf16(false, a0.v, false, b1.v,
                                                    (short)0, acc01, false, false);
    acc10 = __builtin_amdgcn_wmma_f32_16x16x32_bf16(false, a1.v, false, b0.v,
                                                    (short)0, acc10, false, false);
    acc11 = __builtin_amdgcn_wmma_f32_16x16x32_bf16(false, a1.v, false, b1.v,
                                                    (short)0, acc11, false, false);
    if (kt + 1 < nk) {
      wait_async0();                   // next buffer resident (own wave's copies)
      __syncthreads();                 // all waves' copies visible + reads done
    }
  }

  // store: D VGPR m -> row (base + m + hi*8), col base + (lane&15)
  const int cm = tm + wy * 32 + hi * 8;
  const int cn0 = tn + wx * 32 + r;
  const int cn1 = cn0 + 16;
  const float bv0 = bias ? bias[cn0] : 0.0f;
  const float bv1 = bias ? bias[cn1] : 0.0f;
  float* c0 = C + (size_t)cm * N;
#pragma unroll
  for (int m = 0; m < 8; ++m) {
    c0[(size_t)m * N + cn0] = acc00[m] + bv0;
    c0[(size_t)m * N + cn1] = acc01[m] + bv1;
    c0[(size_t)(m + 16) * N + cn0] = acc10[m] + bv0;
    c0[(size_t)(m + 16) * N + cn1] = acc11[m] + bv1;
  }
}

// ---------------------------------------------------------------------------
// Simple GEMM (fallback for small N, e.g. w_out N=64): wave -> one 16x16 tile.
// ---------------------------------------------------------------------------
__global__ void k_gemm_bf16(const unsigned short* __restrict__ A,
                            const unsigned short* __restrict__ Bt,
                            const float* __restrict__ bias,
                            float* __restrict__ C, int M, int N, int K) {
  const int wave = threadIdx.x >> 5;
  const int lane = threadIdx.x & 31;
  const int tm = blockIdx.y * 32 + (wave >> 1) * 16;
  const int tn = blockIdx.x * 32 + (wave & 1) * 16;
  const int r  = lane & 15;
  const int hi = lane >> 4;

  const unsigned short* arow = A  + (size_t)(tm + r) * K;
  const unsigned short* brow = Bt + (size_t)(tn + r) * K;

  v8f acc = {};
  for (int k = 0; k < K; k += 32) {
    FragU a, b;
    a.u[0] = *(const uint4*)(arow + k + hi * 8);
    a.u[1] = *(const uint4*)(arow + k + 16 + hi * 8);
    const uint4* bp = (const uint4*)(brow + k + hi * 8);
    b.u[0] = bp[0];
    b.u[1] = bp[2];
    if (k + 32 < K) {
      __builtin_prefetch(arow + k + 32, 0, 0);
      __builtin_prefetch(brow + k + 32, 0, 0);
    }
    acc = __builtin_amdgcn_wmma_f32_16x16x32_bf16(false, a.v, false, b.v,
                                                  (short)0, acc, false, false);
  }
  const int cn = tn + r;
  const float bv = bias ? bias[cn] : 0.0f;
  float* crow = C + (size_t)(tm + hi * 8) * N + cn;
#pragma unroll
  for (int m = 0; m < 8; ++m)
    crow[(size_t)m * N] = acc[m] + bv;
}

// Weight f32 [K,N] -> bf16 transposed [N,K]
__global__ void k_convt(const float* __restrict__ W, unsigned short* __restrict__ Wt,
                        int K, int N) {
  int idx = blockIdx.x * 256 + threadIdx.x;
  if (idx >= K * N) return;
  int n = idx / K, k = idx - n * K;
  Wt[idx] = f2bf(W[(size_t)k * N + n]);
}

__global__ void k_f32_bf16(const float* __restrict__ x, unsigned short* __restrict__ y, int n) {
  int i = blockIdx.x * 256 + threadIdx.x;
  if (i < n) y[i] = f2bf(x[i]);
}

__global__ void k_fill_zero(float* p, int n) {
  int i = blockIdx.x * 256 + threadIdx.x;
  if (i < n) p[i] = 0.0f;
}

__global__ void k_silu(const float* __restrict__ v, float* __restrict__ o, int n) {
  int i = blockIdx.x * 256 + threadIdx.x;
  if (i < n) { float x = v[i]; o[i] = x / (1.0f + __expf(-x)); }
}

// mod = silu(vec) @ W[768,N] + b
__global__ void k_mod_gemv(const float* __restrict__ sv, const float* __restrict__ W,
                           const float* __restrict__ b, float* __restrict__ out, int N) {
  int j = blockIdx.x * 256 + threadIdx.x;
  if (j >= N) return;
  float acc = b[j];
  for (int k = 0; k < HID; ++k) acc += sv[k] * W[(size_t)k * N + j];
  out[j] = acc;
}

// y = (1+sc)*LN(x) + sh -> bf16
__global__ void k_ln_mod(const float* __restrict__ x, const float* __restrict__ mod,
                         int shOff, int scOff, unsigned short* __restrict__ y) {
  const int row = blockIdx.x;
  const int t = threadIdx.x;
  const float* xr = x + (size_t)row * HID;
  float a0 = xr[t], a1 = xr[t + 256], a2 = xr[t + 512];
  __shared__ float red[256];
  red[t] = a0 + a1 + a2;
  __syncthreads();
  for (int o = 128; o > 0; o >>= 1) { if (t < o) red[t] += red[t + o]; __syncthreads(); }
  const float mean = red[0] * (1.0f / 768.0f);
  __syncthreads();
  float d0 = a0 - mean, d1 = a1 - mean, d2 = a2 - mean;
  red[t] = d0 * d0 + d1 * d1 + d2 * d2;
  __syncthreads();
  for (int o = 128; o > 0; o >>= 1) { if (t < o) red[t] += red[t + o]; __syncthreads(); }
  const float inv = rsqrtf(red[0] * (1.0f / 768.0f) + 1e-6f);
  unsigned short* yr = y + (size_t)row * HID;
  yr[t]       = f2bf((1.0f + mod[scOff + t])       * d0 * inv + mod[shOff + t]);
  yr[t + 256] = f2bf((1.0f + mod[scOff + t + 256]) * d1 * inv + mod[shOff + t + 256]);
  yr[t + 512] = f2bf((1.0f + mod[scOff + t + 512]) * d2 * inv + mod[shOff + t + 512]);
}

// split heads, RMS-norm q/k, RoPE; Q/K/V as [H][SEQ][64] f32
__global__ void k_qkv_prep(const float* __restrict__ qkvT, const float* __restrict__ qkvI,
                           const float* __restrict__ pe,
                           const float* __restrict__ qsT, const float* __restrict__ ksT,
                           const float* __restrict__ qsI, const float* __restrict__ ksI,
                           float* __restrict__ Qb, float* __restrict__ Kb,
                           float* __restrict__ Vb) {
  const int gid  = blockIdx.x * 8 + (threadIdx.x >> 5);
  const int lane = threadIdx.x & 31;
  const int h = gid / SEQ;
  const int s = gid - h * SEQ;
  const float *src, *qs, *ks;
  if (s < TXT) { src = qkvT + (size_t)s * 3 * HID;        qs = qsT; ks = ksT; }
  else         { src = qkvI + (size_t)(s - TXT) * 3 * HID; qs = qsI; ks = ksI; }
  const int d0 = 2 * lane, d1 = 2 * lane + 1;
  float q0 = src[h * HD + d0],           q1 = src[h * HD + d1];
  float k0 = src[HID + h * HD + d0],     k1 = src[HID + h * HD + d1];
  float v0 = src[2 * HID + h * HD + d0], v1 = src[2 * HID + h * HD + d1];
  float qr = rsqrtf(wave_sum(q0 * q0 + q1 * q1) * (1.0f / HD) + 1e-6f);
  float kr = rsqrtf(wave_sum(k0 * k0 + k1 * k1) * (1.0f / HD) + 1e-6f);
  q0 *= qr * qs[d0]; q1 *= qr * qs[d1];
  k0 *= kr * ks[d0]; k1 *= kr * ks[d1];
  const float* pb = pe + ((size_t)s * 32 + lane) * 4;
  float Q0 = pb[0] * q0 + pb[1] * q1, Q1 = pb[2] * q0 + pb[3] * q1;
  float K0 = pb[0] * k0 + pb[1] * k1, K1 = pb[2] * k0 + pb[3] * k1;
  float* qd = Qb + ((size_t)h * SEQ + s) * HD;
  float* kd = Kb + ((size_t)h * SEQ + s) * HD;
  float* vd = Vb + ((size_t)h * SEQ + s) * HD;
  qd[d0] = Q0; qd[d1] = Q1;
  kd[d0] = K0; kd[d1] = K1;
  vd[d0] = v0; vd[d1] = v1;
}

// online-softmax attention; wave per (head, query); out bf16 [SEQ, HID]
__global__ void k_attention(const float* __restrict__ Qb, const float* __restrict__ Kb,
                            const float* __restrict__ Vb, unsigned short* __restrict__ out) {
  const int gid  = blockIdx.x * 8 + (threadIdx.x >> 5);
  const int lane = threadIdx.x & 31;
  const int h = gid / SEQ;
  const int s = gid - h * SEQ;
  const float* q = Qb + ((size_t)h * SEQ + s) * HD;
  const float q0 = q[lane], q1 = q[lane + 32];
  const float* kh = Kb + (size_t)h * SEQ * HD;
  const float* vh = Vb + (size_t)h * SEQ * HD;
  float m = -3.4e38f, l = 0.0f, a0 = 0.0f, a1 = 0.0f;
  for (int j = 0; j < SEQ; ++j) {
    const float* kp = kh + (size_t)j * HD;
    float p = wave_sum(q0 * kp[lane] + q1 * kp[lane + 32]) * 0.125f;
    float mn = fmaxf(m, p);
    float c = __expf(m - mn), w = __expf(p - mn);
    const float* vp = vh + (size_t)j * HD;
    l  = l * c + w;
    a0 = a0 * c + w * vp[lane];
    a1 = a1 * c + w * vp[lane + 32];
    m = mn;
  }
  unsigned short* o = out + (size_t)s * HID + h * HD;
  const float il = 1.0f / l;
  o[lane]      = f2bf(a0 * il);
  o[lane + 32] = f2bf(a1 * il);
}

__global__ void k_gelu_bf16(const float* __restrict__ x, unsigned short* __restrict__ y, int n) {
  int i = blockIdx.x * 256 + threadIdx.x;
  if (i >= n) return;
  float v = x[i];
  float t = tanhf(0.7978845608028654f * (v + 0.044715f * v * v * v));
  y[i] = f2bf(0.5f * v * (1.0f + t));
}

__global__ void k_residual_gate(float* __restrict__ x, const float* __restrict__ y,
                                const float* __restrict__ mod, int gOff, int n) {
  int i = blockIdx.x * 256 + threadIdx.x;
  if (i >= n) return;
  x[i] += mod[gOff + (i % HID)] * y[i];
}

// cond (16,64,64) f32 -> tokens [1024,64] bf16
__global__ void k_embed_tokens(const float* __restrict__ img, unsigned short* __restrict__ A) {
  int i = blockIdx.x * 256 + threadIdx.x;
  int t = i >> 6, f = i & 63;
  int c = f >> 2, dy = (f >> 1) & 1, dx = f & 1;
  int iy = t >> 5, ix = t & 31;
  A[i] = f2bf(img[c * 4096 + (2 * iy + dy) * 64 + (2 * ix + dx)]);
}

// y [1024,64] -> image (16,64,64)
__global__ void k_tokens_to_image(const float* __restrict__ y, float* __restrict__ img) {
  int i = blockIdx.x * 256 + threadIdx.x;
  int c = i >> 12, rem = i & 4095, yy = rem >> 6, xx = rem & 63;
  int t = (yy >> 1) * 32 + (xx >> 1);
  int f = c * 4 + (yy & 1) * 2 + (xx & 1);
  img[i] = y[t * 64 + f];
}

// image_into_patch(norm=True): per (patch, channel) over 256 pixels, ddof=1
__global__ void k_patchify_norm(const float* __restrict__ img, float* __restrict__ out) {
  int p = blockIdx.x >> 4, c = blockIdx.x & 15;
  int q = threadIdx.x, py = q >> 4, px = q & 15;
  int ph = p >> 2, pw = p & 3;
  float v = img[c * 4096 + (ph * 16 + py) * 64 + pw * 16 + px];
  __shared__ float red[256];
  red[q] = v; __syncthreads();
  for (int o = 128; o > 0; o >>= 1) { if (q < o) red[q] += red[q + o]; __syncthreads(); }
  float mean = red[0] * (1.0f / 256.0f);
  __syncthreads();
  float d = v - mean;
  red[q] = d * d; __syncthreads();
  for (int o = 128; o > 0; o >>= 1) { if (q < o) red[q] += red[q + o]; __syncthreads(); }
  float sd = sqrtf(red[0] * (1.0f / 255.0f));
  out[p * 4096 + q * 16 + c] = d / (sd + 1e-6f);
}

// ori[p][type][q*16+c] from the ORIGINAL conds
__global__ void k_patchify_raw(const float* __restrict__ c0, const float* __restrict__ c1,
                               float* __restrict__ ori) {
  int i = blockIdx.x * 256 + threadIdx.x;
  int t = i >> 16;
  int rem = i & 65535;
  int p = rem >> 12, q = (rem >> 4) & 255, c = rem & 15;
  const float* src = t ? c1 : c0;
  int ph = p >> 2, pw = p & 3, py = q >> 4, px = q & 15;
  ori[((size_t)p * 2 + t) * 4096 + q * 16 + c] =
      src[c * 4096 + (ph * 16 + py) * 64 + pw * 16 + px];
}

// scores[p][t] = sum_e (patch_embed[t][p][e] + comb[p][e])
__global__ void k_scores(const float* __restrict__ peb, const float* __restrict__ comb,
                         float* __restrict__ scores) {
  int p = blockIdx.x >> 1, t = blockIdx.x & 1;
  int tid = threadIdx.x;
  const float* pr = peb + ((size_t)t * NPATCH + p) * 4096;
  const float* cr = comb + (size_t)p * 4096;
  float s = 0.0f;
  for (int e = tid; e < 4096; e += 256) s += pr[e] + cr[e];
  __shared__ float red[256];
  red[tid] = s; __syncthreads();
  for (int o = 128; o > 0; o >>= 1) { if (tid < o) red[tid] += red[tid + o]; __syncthreads(); }
  if (tid == 0) scores[p * 2 + t] = red[0];
}

__global__ void k_select(const float* __restrict__ scores, int* __restrict__ zero_mask,
                         int* __restrict__ sel_type, int* __restrict__ upd) {
  if (threadIdx.x != 0) return;
  float mp[NPATCH];
  for (int p = 0; p < NPATCH; ++p) {
    float s0 = scores[2 * p], s1 = scores[2 * p + 1];
    float n = fmaxf(sqrtf(s0 * s0 + s1 * s1), 1e-12f);
    float a = s0 / n, b = s1 / n;
    float mx = fmaxf(a, b);
    float e0 = __expf(a - mx), e1 = __expf(b - mx);
    float inv = 1.0f / (e0 + e1);
    float p0 = e0 * inv, p1 = e1 * inv;
    if (!zero_mask[p]) { p0 = 0.0f; p1 = 0.0f; }
    mp[p] = fmaxf(p0, p1);
    sel_type[p] = (p0 >= p1) ? 0 : 1;
  }
  int taken[NPATCH];
  for (int p = 0; p < NPATCH; ++p) taken[p] = 0;
  for (int r = 0; r < SELN; ++r) {
    int best = 0; float bv = -1.0f;
    for (int p = 0; p < NPATCH; ++p)
      if (!taken[p] && mp[p] > bv) { bv = mp[p]; best = p; }
    taken[best] = 1;
  }
  for (int p = 0; p < NPATCH; ++p) {
    upd[p] = taken[p];
    if (taken[p]) zero_mask[p] = 0;
  }
}

__global__ void k_gather(const float* __restrict__ ori, const int* __restrict__ sel_type,
                         const int* __restrict__ upd, float* __restrict__ sel) {
  int i = blockIdx.x * 256 + threadIdx.x;
  int p = i >> 12;
  float v = ori[((size_t)p * 2 + sel_type[p]) * 4096 + (i & 4095)];
  sel[i] = upd[p] ? v * 2.0f : v;
}

__global__ void k_zero_conds(float* __restrict__ c0, float* __restrict__ c1,
                             const int* __restrict__ upd) {
  int i = blockIdx.x * 256 + threadIdx.x;
  int yy = (i >> 6) & 63, xx = i & 63;
  int p = (yy >> 4) * 4 + (xx >> 4);
  if (upd[p]) { c0[i] = 0.0f; c1[i] = 0.0f; }
}

__global__ void k_unpatchify(const float* __restrict__ sel, float* __restrict__ img) {
  int i = blockIdx.x * 256 + threadIdx.x;
  int c = i >> 12, rem = i & 4095, yy = rem >> 6, xx = rem & 63;
  int p = (yy >> 4) * 4 + (xx >> 4);
  int q = (yy & 15) * 16 + (xx & 15);
  img[i] = sel[p * 4096 + q * 16 + c];
}

__global__ void k_init_state(int* mask) {
  if (threadIdx.x < NPATCH) mask[threadIdx.x] = 1;
}

// ---------------------------------------------------------------------------
// Host orchestration
// ---------------------------------------------------------------------------
struct BW {
  const float *mw_i, *mb_i, *mw_t, *mb_t;
  const float *qkv_i, *qs_i, *ks_i;
  const float *pw, *pb;
  const float *qkv_t, *qs_t, *ks_t;
  const float *w1, *b1, *w2, *b2;
};

struct Ctx {
  hipStream_t s;
  const float *txt, *pe;
  unsigned short *wt, *abuf, *attnb;
  float *qkv_img, *qkv_txt, *Qb, *Kb, *Vb, *hid, *tmp, *imgtok, *ytok, *imgbuf;
  float *sv, *modi, *modt, *scores;
};

static void run_gemm(const Ctx& c, const unsigned short* A, const float* W,
                     int K, int N, int M, const float* bias, float* C) {
  int total = K * N;
  k_convt<<<(total + 255) / 256, 256, 0, c.s>>>(W, c.wt, K, N);
  if ((M % 64 == 0) && (N % 128 == 0)) {
    dim3 grid(N / 128, M / 64);
    k_gemm_bf16_lds<<<grid, 256, 0, c.s>>>(A, c.wt, bias, C, M, N, K);
  } else {
    dim3 grid(N / 32, M / 32);
    k_gemm_bf16<<<grid, 128, 0, c.s>>>(A, c.wt, bias, C, M, N, K);
  }
}

static void run_stream_block(const Ctx& c, const BW& b, float* imgtok) {
  k_mod_gemv<<<(6 * HID) / 256, 256, 0, c.s>>>(c.sv, b.mw_i, b.mb_i, c.modi, 6 * HID);
  k_mod_gemv<<<(6 * HID) / 256, 256, 0, c.s>>>(c.sv, b.mw_t, b.mb_t, c.modt, 6 * HID);
  // img qkv
  k_ln_mod<<<IMG, 256, 0, c.s>>>(imgtok, c.modi, 0, HID, c.abuf);
  run_gemm(c, c.abuf, b.qkv_i, HID, 3 * HID, IMG, nullptr, c.qkv_img);
  // txt qkv
  k_ln_mod<<<TXT, 256, 0, c.s>>>(c.txt, c.modt, 0, HID, c.abuf);
  run_gemm(c, c.abuf, b.qkv_t, HID, 3 * HID, TXT, nullptr, c.qkv_txt);
  // heads + RMS + RoPE
  k_qkv_prep<<<(HEADS * SEQ) / 8, 256, 0, c.s>>>(c.qkv_txt, c.qkv_img, c.pe,
      b.qs_t, b.ks_t, b.qs_i, b.ks_i, c.Qb, c.Kb, c.Vb);
  k_attention<<<(HEADS * SEQ) / 8, 256, 0, c.s>>>(c.Qb, c.Kb, c.Vb, c.attnb);
  // projection on the img half of the attention output
  run_gemm(c, c.attnb + (size_t)TXT * HID, b.pw, HID, HID, IMG, b.pb, c.tmp);
  k_residual_gate<<<(IMG * HID) / 256, 256, 0, c.s>>>(imgtok, c.tmp, c.modi, 2 * HID, IMG * HID);
  // MLP
  k_ln_mod<<<IMG, 256, 0, c.s>>>(imgtok, c.modi, 3 * HID, 4 * HID, c.abuf);
  run_gemm(c, c.abuf, b.w1, HID, MLPD, IMG, b.b1, c.hid);
  k_gelu_bf16<<<(IMG * MLPD) / 256, 256, 0, c.s>>>(c.hid, c.abuf, IMG * MLPD);
  run_gemm(c, c.abuf, b.w2, MLPD, HID, IMG, b.b2, c.tmp);
  k_residual_gate<<<(IMG * HID) / 256, 256, 0, c.s>>>(imgtok, c.tmp, c.modi, 5 * HID, IMG * HID);
}

static void run_branch(const Ctx& c, const BW& b, const float* img_src,
                       const float* w_in, const float* b_in,
                       const float* w_out, const float* b_out, float* patch_out) {
  k_embed_tokens<<<256, 256, 0, c.s>>>(img_src, c.abuf);
  run_gemm(c, c.abuf, w_in, 64, HID, IMG, b_in, c.imgtok);
  run_stream_block(c, b, c.imgtok);
  k_f32_bf16<<<(IMG * HID) / 256, 256, 0, c.s>>>(c.imgtok, c.abuf, IMG * HID);
  run_gemm(c, c.abuf, w_out, HID, 64, IMG, b_out, c.ytok);
  k_tokens_to_image<<<256, 256, 0, c.s>>>(c.ytok, c.imgbuf);
  k_patchify_norm<<<256, 256, 0, c.s>>>(c.imgbuf, patch_out);
}

extern "C" void kernel_launch(void* const* d_in, const int* in_sizes, int n_in,
                              void* d_out, int out_size, void* d_ws, size_t ws_size,
                              hipStream_t stream) {
  (void)in_sizes; (void)n_in; (void)out_size; (void)ws_size;

  const float* cond_canny = (const float*)d_in[0];
  const float* cond_depth = (const float*)d_in[1];
  const float* txt  = (const float*)d_in[2];
  const float* pe   = (const float*)d_in[3];
  const float* vec  = (const float*)d_in[4];
  const float* w_in  = (const float*)d_in[5];
  const float* b_in  = (const float*)d_in[6];
  const float* w_out = (const float*)d_in[7];
  const float* b_out = (const float*)d_in[8];

  auto getBW = [&](int bi) -> BW {
    int o = 9 + 16 * bi;
    BW b;
    b.mw_i  = (const float*)d_in[o + 0];  b.mb_i = (const float*)d_in[o + 1];
    b.mw_t  = (const float*)d_in[o + 2];  b.mb_t = (const float*)d_in[o + 3];
    b.qkv_i = (const float*)d_in[o + 4];
    b.qs_i  = (const float*)d_in[o + 5];  b.ks_i = (const float*)d_in[o + 6];
    b.pw    = (const float*)d_in[o + 7];  b.pb   = (const float*)d_in[o + 8];
    b.qkv_t = (const float*)d_in[o + 9];
    b.qs_t  = (const float*)d_in[o + 10]; b.ks_t = (const float*)d_in[o + 11];
    b.w1    = (const float*)d_in[o + 12]; b.b1   = (const float*)d_in[o + 13];
    b.w2    = (const float*)d_in[o + 14]; b.b2   = (const float*)d_in[o + 15];
    return b;
  };
  BW blocks[3] = { getBW(0), getBW(1), getBW(2) };

  // ---- workspace carve-up (256B aligned) ----
  char* base = (char*)d_ws;
  size_t off = 0;
  auto alloc = [&](size_t bytes) -> void* {
    void* p = base + off;
    off = (off + bytes + 255) & ~(size_t)255;
    return p;
  };

  Ctx c;
  c.s   = stream;
  c.txt = txt;
  c.pe  = pe;
  c.wt      = (unsigned short*)alloc((size_t)HID * 6 * HID * 2);
  c.abuf    = (unsigned short*)alloc((size_t)IMG * MLPD * 2);
  c.attnb   = (unsigned short*)alloc((size_t)SEQ * HID * 2);
  c.qkv_img = (float*)alloc((size_t)IMG * 3 * HID * 4);
  c.qkv_txt = (float*)alloc((size_t)TXT * 3 * HID * 4);
  c.Qb      = (float*)alloc((size_t)HEADS * SEQ * HD * 4);
  c.Kb      = (float*)alloc((size_t)HEADS * SEQ * HD * 4);
  c.Vb      = (float*)alloc((size_t)HEADS * SEQ * HD * 4);
  c.hid     = (float*)alloc((size_t)IMG * MLPD * 4);
  c.tmp     = (float*)alloc((size_t)IMG * HID * 4);
  c.imgtok  = (float*)alloc((size_t)IMG * HID * 4);
  c.ytok    = (float*)alloc((size_t)IMG * 64 * 4);
  c.imgbuf  = (float*)alloc((size_t)65536 * 4);
  c.sv      = (float*)alloc(HID * 4);
  c.modi    = (float*)alloc(6 * HID * 4);
  c.modt    = (float*)alloc(6 * HID * 4);
  c.scores  = (float*)alloc(2 * NPATCH * 4);

  float* condw0   = (float*)alloc((size_t)65536 * 4);
  float* condw1   = (float*)alloc((size_t)65536 * 4);
  float* ori      = (float*)alloc((size_t)NPATCH * 2 * 4096 * 4);
  float* pebuf    = (float*)alloc((size_t)2 * NPATCH * 4096 * 4);
  float* combp    = (float*)alloc((size_t)NPATCH * 4096 * 4);
  float* selected = (float*)alloc((size_t)NPATCH * 4096 * 4);
  float* selimg   = (float*)alloc((size_t)65536 * 4);
  int*   mask     = (int*)alloc(NPATCH * 4);
  int*   seltype  = (int*)alloc(NPATCH * 4);
  int*   upd      = (int*)alloc(NPATCH * 4);

  // ---- init (deterministic every call) ----
  hipMemcpyAsync(condw0, cond_canny, (size_t)65536 * 4, hipMemcpyDeviceToDevice, stream);
  hipMemcpyAsync(condw1, cond_depth, (size_t)65536 * 4, hipMemcpyDeviceToDevice, stream);
  k_patchify_raw<<<512, 256, 0, stream>>>(cond_canny, cond_depth, ori);
  k_fill_zero<<<256, 256, 0, stream>>>(selimg, 65536);
  k_silu<<<3, 256, 0, stream>>>(vec, c.sv, HID);
  k_init_state<<<1, 32, 0, stream>>>(mask);

  // ---- main selection loop ----
  for (int it = 0; it < NITER; ++it) {
    run_branch(c, blocks[0], condw0, w_in, b_in, w_out, b_out, pebuf);
    run_branch(c, blocks[1], condw1, w_in, b_in, w_out, b_out, pebuf + NPATCH * 4096);
    run_branch(c, blocks[2], selimg, w_in, b_in, w_out, b_out, combp);

    k_scores<<<2 * NPATCH, 256, 0, stream>>>(pebuf, combp, c.scores);
    k_select<<<1, 32, 0, stream>>>(c.scores, mask, seltype, upd);
    k_gather<<<256, 256, 0, stream>>>(ori, seltype, upd, selected);
    k_zero_conds<<<256, 256, 0, stream>>>(condw0, condw1, upd);
    k_unpatchify<<<256, 256, 0, stream>>>(selected, selimg);
  }

  hipMemcpyAsync(d_out, selimg, (size_t)65536 * 4, hipMemcpyDeviceToDevice, stream);
}
